// RealNVP_LiteFiLM_8211977470787
// MI455X (gfx1250) — matrix operand order repaired and checked by reading, not compiled, not run on hardware
//
#include <hip/hip_runtime.h>
#include <hip/hip_bf16.h>
#include <stdint.h>

typedef __attribute__((ext_vector_type(16))) _Float16 v16h;
typedef __attribute__((ext_vector_type(8)))  float    v8f;

namespace {

constexpr int NW  = 8;          // waves per block (wave32)
constexpr int RPW = 16;         // rows per wave (WMMA M)
constexpr int RPB = NW * RPW;   // 128 rows per block
constexpr int L = 8;
constexpr int WST = 40;         // f16 weight-tile row stride (halves); 80B, 16B-aligned

// small-param LDS offsets (floats)
enum {
  P_embW1 = 0,   P_embb1 = 128, P_embb2 = 160, P_tW  = 192,
  P_tb  = 320,   P_tlng  = 352, P_tlnb  = 384, P_tgb = 416,
  P_tbb = 448,   P_tob   = 480, P_sW    = 484, P_sb  = 612,
  P_slng = 644,  P_slnb  = 676, P_sgb   = 708, P_sbb = 740,
  P_sob = 772,
  P_tS  = 776,   // 9 LN-stat scalars, t path
  P_sS  = 785,   // 9 LN-stat scalars, s path
  P_TOTAL = 794
};

struct Params {
  const float *x, *y;
  const float *embW1, *embb1, *embW2, *embb2;
  const float *tW, *tb, *t_ln_g, *t_ln_b, *t_gW, *t_gb, *t_bW, *t_bb, *toW, *tob;
  const float *sW, *sb, *s_ln_g, *s_ln_b, *s_gW, *s_gb, *s_bW, *s_bb, *soW, *sob;
  float *out_z, *out_ld;
};

union V16H { v16h v; uint32_t u[8]; _Float16 h[16]; uint4 q[2]; };

__device__ __forceinline__ v8f wmma_f16(v16h a, v16h b) {
  v8f c = {0.f, 0.f, 0.f, 0.f, 0.f, 0.f, 0.f, 0.f};
  // (neg_a, A, neg_b, B, c_mod, C, reuse_a, reuse_b)
  return __builtin_amdgcn_wmma_f32_16x16x32_f16(false, a, false, b, (short)0, c, false, false);
}

// K index of A-operand slot (v, p) for this half-wave (16-bit 16x32 A layout)
__device__ __forceinline__ int aK(int v, int half) {
  const int kb = half ? 8 : 0;
  return (v < 4) ? (kb + 2 * v) : (16 + kb + 2 * (v - 4));
}

// A operand from a 16x32 f16 row-major LDS tile: two contiguous 16B runs per lane
__device__ __forceinline__ v16h ldsA(const _Float16* et, int lq, int half) {
  const int kb = half ? 8 : 0;
  V16H a;
  a.q[0] = *(const uint4*)(et + lq * 32 + kb);        // k = kb .. kb+7
  a.q[1] = *(const uint4*)(et + lq * 32 + 16 + kb);   // k = 16+kb .. 16+kb+7
  return a.v;
}

// B operand: 32x16 tile of a 32x32 f16 weight stored transposed [n][k], stride WST
// lanes 0-15: N=lane, K=0..7,16..23; lanes 16-31: N=lane-16, K=8..15,24..31 -> two 16B runs
__device__ __forceinline__ v16h ldsB(const _Float16* wt, int lq, int half, int nt) {
  const int n = nt * 16 + lq;
  const _Float16* bp = wt + n * WST + (half ? 8 : 0);
  V16H b;
  b.q[0] = *(const uint4*)(bp);
  b.q[1] = *(const uint4*)(bp + 16);
  return b.v;
}

__device__ __forceinline__ void cpyP(float* dst, const float* src, int n, int tid) {
  for (int t = tid; t < n; t += 256) dst[t] = src[t];
}

// Dense(masked dims) -> LayerNorm (closed-form stats) -> FiLM -> relu, f16 out
__device__ __forceinline__ void film_path(
    const float* sp, const float (&zr)[8][2], _Float16* eout,
    int oW, int ob, int olng, int olnb, int oS,
    const float (&gm)[8][2], const float (&be)[8][2],
    int lq, int half, int dm0)
{
  const int n0 = lq, n1 = lq + 16;
  const float w00 = sp[oW + dm0 * 32 + n0],       w01 = sp[oW + dm0 * 32 + n1];
  const float w10 = sp[oW + (dm0 + 1) * 32 + n0], w11 = sp[oW + (dm0 + 1) * 32 + n1];
  const float bb0 = sp[ob + n0],   bb1 = sp[ob + n1];
  const float lg0 = sp[olng + n0], lg1 = sp[olng + n1];
  const float lb0 = sp[olnb + n0], lb1 = sp[olnb + n1];
  const float Mw0 = sp[oS + 0], Mw1 = sp[oS + 1], Mb  = sp[oS + 2];
  const float S00 = sp[oS + 3], S11 = sp[oS + 4], S01 = sp[oS + 5];
  const float S0b = sp[oS + 6], S1b = sp[oS + 7], Sbb = sp[oS + 8];
#pragma unroll
  for (int v = 0; v < 8; ++v) {
    const int m = half * 8 + v;
    const float z0 = zr[v][0], z1 = zr[v][1];
    const float h0 = bb0 + z0 * w00 + z1 * w10;
    const float h1 = bb1 + z0 * w01 + z1 * w11;
    // exact column sums of h and h^2 as quadratic forms in (z0, z1)
    const float mean = (Mw0 * z0 + Mw1 * z1 + Mb) * (1.f / 32.f);
    const float sq   = S00 * z0 * z0 + S11 * z1 * z1 + Sbb
                     + 2.f * (S01 * z0 * z1 + S0b * z0 + S1b * z1);
    const float var  = sq * (1.f / 32.f) - mean * mean;
    const float inv  = rsqrtf(var + 1e-5f);
    const float t0v = (h0 - mean) * inv * lg0 + lb0;
    const float t1v = (h1 - mean) * inv * lg1 + lb1;
    eout[m * 32 + n0] = (_Float16)fmaxf((1.f + gm[v][0]) * t0v + be[v][0], 0.f);
    eout[m * 32 + n1] = (_Float16)fmaxf((1.f + gm[v][1]) * t1v + be[v][1], 0.f);
  }
}

} // namespace

__global__ void __launch_bounds__(256)
realnvp_film_wmma(Params p) {
  // [0]=embW2 [1]=t_gW [2]=t_bW [3]=s_gW [4]=s_bW [5]=toW(pad) [6]=soW(pad)
  __shared__ alignas(16) _Float16 sW16[7][32 * WST];
  __shared__ alignas(16) float    sP[P_TOTAL];
  __shared__ alignas(16) _Float16 sE[NW][RPW * 32];  // e2/t_h/s_h f16 staging
  __shared__ alignas(16) float    sZ[NW][RPW * 4];   // evolving z per wave
  __shared__ alignas(16) float    sYt[NW][RPW * 4];  // y per wave
  __shared__ alignas(16) float    sTF[NW][RPW * 4];  // t_final
  __shared__ alignas(16) float    sSF[NW][RPW * 4];  // s_final (post tanh)

  const int tid  = threadIdx.x;
  const int w    = tid >> 5;
  const int lane = tid & 31;
  const int lq   = lane & 15;
  const int half = lane >> 4;
  const long rowBase = (long)blockIdx.x * RPB + (long)w * RPW;

  // zero the pad rows (n=4..15) of the two 32x4 projection B tiles, once
  for (int g = tid; g < 12 * 32; g += 256) {
    const int n = 4 + (g >> 5), k = g & 31;
    sW16[5][n * WST + k] = (_Float16)0.f;
    sW16[6][n * WST + k] = (_Float16)0.f;
  }

  if (half == 0) {
    *(float4*)&sZ[w][lq * 4]  = *(const float4*)(p.x + (rowBase + lq) * 4);
    *(float4*)&sYt[w][lq * 4] = *(const float4*)(p.y + (rowBase + lq) * 4);
  }
  float ldacc = 0.f;
  const int m2 = lane >> 1;          // row for the coupling update
  const int d0 = (lane & 1) * 2;     // dim pair

#pragma unroll 1
  for (int i = 0; i < L; ++i) {
    const int dm0 = (i & 1) ? 2 : 0;   // masked (input) dim pair
    const int df0 = (i & 1) ? 0 : 2;   // free (transformed) dim pair

    __syncthreads();                   // protect prior-layer reads of sP/sW16
    // ---- cooperative staging of per-layer parameters into LDS ----
    cpyP(sP + P_embW1, p.embW1 + i * 128, 128, tid);
    cpyP(sP + P_embb1, p.embb1 + i * 32, 32, tid);
    cpyP(sP + P_embb2, p.embb2 + i * 32, 32, tid);
    cpyP(sP + P_tW,   p.tW   + i * 128, 128, tid);
    cpyP(sP + P_tb,   p.tb   + i * 32, 32, tid);
    cpyP(sP + P_tlng, p.t_ln_g + i * 32, 32, tid);
    cpyP(sP + P_tlnb, p.t_ln_b + i * 32, 32, tid);
    cpyP(sP + P_tgb,  p.t_gb + i * 32, 32, tid);
    cpyP(sP + P_tbb,  p.t_bb + i * 32, 32, tid);
    cpyP(sP + P_tob,  p.tob  + i * 4, 4, tid);
    cpyP(sP + P_sW,   p.sW   + i * 128, 128, tid);
    cpyP(sP + P_sb,   p.sb   + i * 32, 32, tid);
    cpyP(sP + P_slng, p.s_ln_g + i * 32, 32, tid);
    cpyP(sP + P_slnb, p.s_ln_b + i * 32, 32, tid);
    cpyP(sP + P_sgb,  p.s_gb + i * 32, 32, tid);
    cpyP(sP + P_sbb,  p.s_bb + i * 32, 32, tid);
    cpyP(sP + P_sob,  p.sob  + i * 4, 4, tid);
    {
      const float* g0 = p.embW2 + i * 1024;
      const float* g1 = p.t_gW  + i * 1024;
      const float* g2 = p.t_bW  + i * 1024;
      const float* g3 = p.s_gW  + i * 1024;
      const float* g4 = p.s_bW  + i * 1024;
      for (int g = tid; g < 1024; g += 256) {
        const int k = g >> 5, n = g & 31, o = n * WST + k;  // transposed [n][k]
        sW16[0][o] = (_Float16)g0[g];
        sW16[1][o] = (_Float16)g1[g];
        sW16[2][o] = (_Float16)g2[g];
        sW16[3][o] = (_Float16)g3[g];
        sW16[4][o] = (_Float16)g4[g];
      }
      const float* g5 = p.toW + i * 128;   // (32,4) -> rows n=0..3 of B tile
      const float* g6 = p.soW + i * 128;
      for (int g = tid; g < 128; g += 256) {
        const int k = g >> 2, n = g & 3, o = n * WST + k;
        sW16[5][o] = (_Float16)g5[g];
        sW16[6][o] = (_Float16)g6[g];
      }
      if (i + 1 < L) {                 // warm L2 for next layer's GEMM weights
        __builtin_prefetch(p.embW2 + (i + 1) * 1024, 0, 1);
        __builtin_prefetch(p.t_gW  + (i + 1) * 1024, 0, 1);
        __builtin_prefetch(p.t_bW  + (i + 1) * 1024, 0, 1);
        __builtin_prefetch(p.s_gW  + (i + 1) * 1024, 0, 1);
        __builtin_prefetch(p.s_bW  + (i + 1) * 1024, 0, 1);
      }
    }
    // ---- LN-stat scalars (9 per path) via one 32-lane butterfly, waves 0/1 ----
    if (w < 2) {
      const float* gW = (w ? p.sW : p.tW) + i * 128;   // from global (L2-hot),
      const float* gB = (w ? p.sb : p.tb) + i * 32;    // independent of sP staging
      const float w0 = gW[dm0 * 32 + lane];
      const float w1 = gW[(dm0 + 1) * 32 + lane];
      const float bb = gB[lane];
      float v9[9] = { w0, w1, bb, w0 * w0, w1 * w1, w0 * w1, w0 * bb, w1 * bb, bb * bb };
#pragma unroll
      for (int k = 0; k < 9; ++k)
#pragma unroll
        for (int d = 1; d <= 16; d <<= 1) v9[k] += __shfl_xor(v9[k], d, 32);
      if (lane == 0) {
        const int oS = w ? P_sS : P_tS;
#pragma unroll
        for (int k = 0; k < 9; ++k) sP[oS + k] = v9[k];
      }
    }
    __syncthreads();

    // ---- hoisted z loads (masked pair) for both FiLM paths ----
    float zr[8][2];
#pragma unroll
    for (int v = 0; v < 8; ++v) {
      const float2 zz = *(const float2*)&sZ[w][(half * 8 + v) * 4 + dm0];
      zr[v][0] = zz.x; zr[v][1] = zz.y;
    }

    // ---- e1 = relu(y @ embW1 + embb1), built directly in A-operand layout ----
    V16H aE;
    {
      const float y0 = sYt[w][lq * 4 + 0], y1 = sYt[w][lq * 4 + 1];
      const float y2 = sYt[w][lq * 4 + 2], y3 = sYt[w][lq * 4 + 3];
#pragma unroll
      for (int v = 0; v < 8; ++v) {
        const int k = aK(v, half);            // this slot's (row=lq, col=k,k+1)
        float a0 = sP[P_embb1 + k]     + y0 * sP[P_embW1 + k]
                 + y1 * sP[P_embW1 + 32 + k] + y2 * sP[P_embW1 + 64 + k]
                 + y3 * sP[P_embW1 + 96 + k];
        float a1 = sP[P_embb1 + k + 1] + y0 * sP[P_embW1 + k + 1]
                 + y1 * sP[P_embW1 + 33 + k] + y2 * sP[P_embW1 + 65 + k]
                 + y3 * sP[P_embW1 + 97 + k];
        aE.h[2 * v]     = (_Float16)fmaxf(a0, 0.f);
        aE.h[2 * v + 1] = (_Float16)fmaxf(a1, 0.f);
      }
    }

    // ---- e2 = relu(e1 @ embW2 + embb2): 2x WMMA ----
    {
      const v8f c0 = wmma_f16(aE.v, ldsB(sW16[0], lq, half, 0));
      const v8f c1 = wmma_f16(aE.v, ldsB(sW16[0], lq, half, 1));
      const float b0 = sP[P_embb2 + lq], b1 = sP[P_embb2 + lq + 16];
      _Float16* et = sE[w];
#pragma unroll
      for (int v = 0; v < 8; ++v) {
        const int m = half * 8 + v;           // C layout: M = v + 8*half, N = lq
        et[m * 32 + lq]      = (_Float16)fmaxf(c0[v] + b0, 0.f);
        et[m * 32 + lq + 16] = (_Float16)fmaxf(c1[v] + b1, 0.f);
      }
    }

    // ---- gamma/beta for t and s: four (16,32)@(32,32) GEMMs = 8x WMMA ----
    float gt[8][2], bt[8][2], gs[8][2], bs[8][2];
    {
      const v16h aC = ldsA(sE[w], lq, half);
      {
        const v8f r0 = wmma_f16(aC, ldsB(sW16[1], lq, half, 0));
        const v8f r1 = wmma_f16(aC, ldsB(sW16[1], lq, half, 1));
        const float b0 = sP[P_tgb + lq], b1 = sP[P_tgb + lq + 16];
#pragma unroll
        for (int v = 0; v < 8; ++v) { gt[v][0] = r0[v] + b0; gt[v][1] = r1[v] + b1; }
      }
      {
        const v8f r0 = wmma_f16(aC, ldsB(sW16[2], lq, half, 0));
        const v8f r1 = wmma_f16(aC, ldsB(sW16[2], lq, half, 1));
        const float b0 = sP[P_tbb + lq], b1 = sP[P_tbb + lq + 16];
#pragma unroll
        for (int v = 0; v < 8; ++v) { bt[v][0] = r0[v] + b0; bt[v][1] = r1[v] + b1; }
      }
      {
        const v8f r0 = wmma_f16(aC, ldsB(sW16[3], lq, half, 0));
        const v8f r1 = wmma_f16(aC, ldsB(sW16[3], lq, half, 1));
        const float b0 = sP[P_sgb + lq], b1 = sP[P_sgb + lq + 16];
#pragma unroll
        for (int v = 0; v < 8; ++v) { gs[v][0] = r0[v] + b0; gs[v][1] = r1[v] + b1; }
      }
      {
        const v8f r0 = wmma_f16(aC, ldsB(sW16[4], lq, half, 0));
        const v8f r1 = wmma_f16(aC, ldsB(sW16[4], lq, half, 1));
        const float b0 = sP[P_sbb + lq], b1 = sP[P_sbb + lq + 16];
#pragma unroll
        for (int v = 0; v < 8; ++v) { bs[v][0] = r0[v] + b0; bs[v][1] = r1[v] + b1; }
      }
    }

    // ---- t path: FiLM -> f16 tile -> (16x32)@(32x4) via padded WMMA ----
    film_path(sP, zr, sE[w], P_tW, P_tb, P_tlng, P_tlnb, P_tS, gt, bt, lq, half, dm0);
    {
      const v16h aT = ldsA(sE[w], lq, half);
      const v8f c = wmma_f16(aT, ldsB(sW16[5], lq, half, 0));
      if (lq < 4) {                       // only output cols 0..3 are real
        const float bo = sP[P_tob + lq];
#pragma unroll
        for (int v = 0; v < 8; ++v) sTF[w][(half * 8 + v) * 4 + lq] = c[v] + bo;
      }
    }

    // ---- s path ----
    film_path(sP, zr, sE[w], P_sW, P_sb, P_slng, P_slnb, P_sS, gs, bs, lq, half, dm0);
    {
      const v16h aS = ldsA(sE[w], lq, half);
      const v8f c = wmma_f16(aS, ldsB(sW16[6], lq, half, 0));
      if (lq < 4) {
        const float bo = sP[P_sob + lq];
#pragma unroll
        for (int v = 0; v < 8; ++v)
          sSF[w][(half * 8 + v) * 4 + lq] = tanhf(c[v] + bo);
      }
    }

    // ---- coupling update: z = z*mask + (z*exp(s)+t)*rmask ; log_det += s*rmask ----
    {
      const float2 tf = *(const float2*)&sTF[w][m2 * 4 + d0];
      const float2 sf = *(const float2*)&sSF[w][m2 * 4 + d0];
      float z0 = sZ[w][m2 * 4 + d0], z1 = sZ[w][m2 * 4 + d0 + 1];
      float ldp = 0.f;
      if (d0 == df0) {
        z0 = z0 * __expf(sf.x) + tf.x;
        z1 = z1 * __expf(sf.y) + tf.y;
        ldp = sf.x + sf.y;
      }
      sZ[w][m2 * 4 + d0]     = z0;
      sZ[w][m2 * 4 + d0 + 1] = z1;
      ldacc += ldp + __shfl_xor(ldp, 1, 32);   // both lanes of a row get the row sum
    }
  }

  // ---- epilogue: z then log_det, concatenated in d_out ----
  {
    const long r = rowBase + m2;
    *(float2*)(p.out_z + r * 4 + d0) = *(const float2*)&sZ[w][m2 * 4 + d0];
    if ((lane & 1) == 0) p.out_ld[r] = ldacc;
  }
}

extern "C" void kernel_launch(void* const* d_in, const int* in_sizes, int n_in,
                              void* d_out, int out_size, void* d_ws, size_t ws_size,
                              hipStream_t stream) {
  (void)n_in; (void)out_size; (void)d_ws; (void)ws_size;
  Params p;
  p.x      = (const float*)d_in[0];
  p.y      = (const float*)d_in[1];
  p.embW1  = (const float*)d_in[2];
  p.embb1  = (const float*)d_in[3];
  p.embW2  = (const float*)d_in[4];
  p.embb2  = (const float*)d_in[5];
  p.tW     = (const float*)d_in[6];
  p.tb     = (const float*)d_in[7];
  p.t_ln_g = (const float*)d_in[8];
  p.t_ln_b = (const float*)d_in[9];
  p.t_gW   = (const float*)d_in[10];
  p.t_gb   = (const float*)d_in[11];
  p.t_bW   = (const float*)d_in[12];
  p.t_bb   = (const float*)d_in[13];
  p.toW    = (const float*)d_in[14];
  p.tob    = (const float*)d_in[15];
  p.sW     = (const float*)d_in[16];
  p.sb     = (const float*)d_in[17];
  p.s_ln_g = (const float*)d_in[18];
  p.s_ln_b = (const float*)d_in[19];
  p.s_gW   = (const float*)d_in[20];
  p.s_gb   = (const float*)d_in[21];
  p.s_bW   = (const float*)d_in[22];
  p.s_bb   = (const float*)d_in[23];
  p.soW    = (const float*)d_in[24];
  p.sob    = (const float*)d_in[25];

  const int B = in_sizes[0] / 4;          // 262144; divisible by 128
  p.out_z  = (float*)d_out;
  p.out_ld = (float*)d_out + (size_t)B * 4;

  const int grid = B / RPB;
  realnvp_film_wmma<<<grid, 256, 0, stream>>>(p);
}
